// BFConv2d_37434934952030
// MI455X (gfx1250) — matrix operand order repaired and checked by reading, not compiled, not run on hardware
//
#include <hip/hip_runtime.h>

typedef __attribute__((ext_vector_type(16))) _Float16 v16h;
typedef __attribute__((ext_vector_type(8)))  _Float16 h8;
typedef __attribute__((ext_vector_type(8)))  float    v8f;
typedef int v4i __attribute__((vector_size(16)));

#define B_    32
#define CIN   256
#define COUT  256
#define H_    56
#define W_    56
#define HP    58      // padded H
#define WP    58      // padded W
#define HW    (H_ * W_)          // 3136
#define NPIX  (B_ * HW)          // 100352
#define G_    8                  // channel groups (256/32)
#define SZ_   32                 // BFP group size
#define QBIT  8

// ---- conv tiling ----
#define MT  128       // cout per block
#define NTP 128       // pixels per block
#define KT  32        // K chunk (cin) per stage
#define WS  40        // LDS row stride (halves), padded -> conflict-free
#define XS  40
#define NK  72        // 9 taps * 8 cin chunks

// ---------------- CDNA5 async-to-LDS support (guarded) ----------------
#if defined(__gfx1250__) && __has_builtin(__builtin_amdgcn_global_load_async_to_lds_b128)
#define HAVE_ASYNC 1
#else
#define HAVE_ASYNC 0
#endif

template <int N>
__device__ __forceinline__ void wait_asynccnt() {
#if defined(__gfx1250__)
#if __has_builtin(__builtin_amdgcn_s_wait_asynccnt)
    __builtin_amdgcn_s_wait_asynccnt(N);
#else
    asm volatile("s_wait_asynccnt %0" ::"n"(N) : "memory");
#endif
#endif
}

#if HAVE_ASYNC
typedef __attribute__((address_space(1))) v4i* g128p;   // global b128 pointer
typedef __attribute__((address_space(3))) v4i* l128p;   // LDS b128 pointer

__device__ __forceinline__ void async_copy_b128(const _Float16* g, _Float16* l) {
    __builtin_amdgcn_global_load_async_to_lds_b128(
        (g128p)(uintptr_t)g,
        (l128p)(unsigned)(uintptr_t)l,
        0, 0);
}
#endif

// ---------------- BFP helpers ----------------
__device__ __forceinline__ int bfp_exp(float absmax) {
    // floor(log2(absmax)) for normal positive floats
    return (int)((__float_as_uint(absmax) >> 23) & 0xFFu) - 127;
}
__device__ __forceinline__ float bfp_quant_val(float v, float iscale, float scale) {
    float m = rintf(v * iscale);                  // round-to-nearest-even
    m = fminf(fmaxf(m, -128.0f), 127.0f);         // clip to [-qmax-1, qmax]
    return m * scale;
}

// ---------------- kernel 1: quantize x -> padded NHWC f16 ----------------
__global__ void __launch_bounds__(256)
quant_x_kernel(const float* __restrict__ x, _Float16* __restrict__ xq) {
    unsigned id = blockIdx.x * blockDim.x + threadIdx.x;   // over B_*G_*HW
    unsigned p  = id % HW;
    unsigned bg = id / HW;
    unsigned g  = bg % G_;
    unsigned b  = bg / G_;

    const float* src = x + (size_t)(b * CIN + g * SZ_) * HW + p;
    float v[SZ_];
    float amax = 0.0f;
#pragma unroll
    for (int c = 0; c < SZ_; ++c) {
        v[c] = src[(size_t)c * HW];
        amax = fmaxf(amax, fabsf(v[c]));
    }
    h8 q[4];
    if (amax > 0.0f) {
        int   e      = bfp_exp(amax);
        float scale  = ldexpf(1.0f, e - (QBIT - 1));
        float iscale = ldexpf(1.0f, (QBIT - 1) - e);
#pragma unroll
        for (int c = 0; c < SZ_; ++c)
            q[c >> 3][c & 7] = (_Float16)bfp_quant_val(v[c], iscale, scale);
    } else {
#pragma unroll
        for (int c = 0; c < SZ_; ++c)
            q[c >> 3][c & 7] = (_Float16)v[c];
    }
    unsigned oh = p / W_, ow = p % W_;
    _Float16* dst = xq + ((size_t)(b * HP + oh + 1) * WP + (ow + 1)) * CIN + g * SZ_;
    *(h8*)(dst)      = q[0];
    *(h8*)(dst + 8)  = q[1];
    *(h8*)(dst + 16) = q[2];
    *(h8*)(dst + 24) = q[3];
}

// ---------------- kernel 2: quantize w -> (tap, cout, cin) f16 ----------------
__global__ void __launch_bounds__(256)
quant_w_kernel(const float* __restrict__ w, _Float16* __restrict__ wq) {
    unsigned id = blockIdx.x * blockDim.x + threadIdx.x;   // over COUT*9*G_
    if (id >= COUT * 9 * G_) return;
    unsigned g   = id % G_;
    unsigned tap = (id / G_) % 9;
    unsigned co  = id / (G_ * 9);

    const float* src = w + ((size_t)(co * CIN + g * SZ_)) * 9 + tap;
    float v[SZ_];
    float amax = 0.0f;
#pragma unroll
    for (int c = 0; c < SZ_; ++c) {
        v[c] = src[c * 9];
        amax = fmaxf(amax, fabsf(v[c]));
    }
    h8 q[4];
    if (amax > 0.0f) {
        int   e      = bfp_exp(amax);
        float scale  = ldexpf(1.0f, e - (QBIT - 1));
        float iscale = ldexpf(1.0f, (QBIT - 1) - e);
#pragma unroll
        for (int c = 0; c < SZ_; ++c)
            q[c >> 3][c & 7] = (_Float16)bfp_quant_val(v[c], iscale, scale);
    } else {
#pragma unroll
        for (int c = 0; c < SZ_; ++c)
            q[c >> 3][c & 7] = (_Float16)v[c];
    }
    _Float16* dst = wq + ((size_t)(tap * COUT + co)) * CIN + g * SZ_;
    *(h8*)(dst)      = q[0];
    *(h8*)(dst + 8)  = q[1];
    *(h8*)(dst + 16) = q[2];
    *(h8*)(dst + 24) = q[3];
}

// ---------------- kernel 3: implicit-GEMM conv via WMMA f16 ----------------
// GEMM: D[M=cout][N=pixel] += A[M][K=cin] * B[K][N]; K = 9 taps x 8 cin chunks.
// Block: 128 cout x 128 pixels, 8 waves, wave tile 64 cout x 32 pixels (4x2 WMMA).
// 2-deep pipeline: async-copy tile k+1 into LDS while computing tile k.
__global__ void __launch_bounds__(256)
conv_wmma_kernel(const _Float16* __restrict__ xq,
                 const _Float16* __restrict__ wq,
                 float* __restrict__ out) {
    __shared__ _Float16 sW[2][MT * WS];    // [buf][cout_local][cin]
    __shared__ _Float16 sX[2][NTP * XS];   // [buf][pixel_local][cin]

    const unsigned tid   = threadIdx.x;
    const unsigned lane  = tid & 31;
    const unsigned wave  = tid >> 5;        // 0..7
    const unsigned waveM = wave & 1;        // cout sub-tile (x64)
    const unsigned waveN = wave >> 1;       // pixel sub-tile (x32)
    const unsigned l16   = lane & 15;
    const unsigned hlf   = lane >> 4;       // 0 or 1

    const unsigned coutBase = blockIdx.y * MT;
    const unsigned pixBase  = blockIdx.x * NTP;

    // staging assignment: thread -> (row 0..127, 16-half segment 0..1) for both tiles
    const unsigned trow = tid >> 1;
    const unsigned tseg = tid & 1;

    // per-thread global bases (fixed across the K loop)
    const unsigned gp = pixBase + trow;
    const unsigned b  = gp / HW;
    const unsigned pp = gp % HW;
    const unsigned oh = pp / W_;
    const unsigned ow = pp % W_;
    const _Float16* xbase = xq + ((size_t)(b * HP + oh) * WP + ow) * CIN + tseg * 16;
    const _Float16* wbase = wq + (size_t)(coutBase + trow) * CIN + tseg * 16;
    const unsigned ldsoff = trow * WS + tseg * 16;   // same stride for both tiles

    v8f acc[4][2];
    const v8f vzero = {0.f, 0.f, 0.f, 0.f, 0.f, 0.f, 0.f, 0.f};
#pragma unroll
    for (int mi = 0; mi < 4; ++mi)
#pragma unroll
        for (int ni = 0; ni < 2; ++ni) acc[mi][ni] = vzero;

    auto src_addrs = [&](int k, const _Float16*& gx, const _Float16*& gw) {
        const int tap = k >> 3, kc = k & 7;
        const int kh = tap / 3, kw = tap % 3;
        gx = xbase + ((size_t)(kh * WP + kw)) * CIN + kc * KT;
        gw = wbase + (size_t)tap * (COUT * CIN) + kc * KT;
    };

    auto compute = [&](int bufi) {
        // A fragments (16 cout x 32 cin): lanes 0-15 hold K 0..7 & 16..23,
        // lanes 16-31 hold K 8..15 & 24..31.
        v16h afr[4];
#pragma unroll
        for (int mi = 0; mi < 4; ++mi) {
            const _Float16* pw = &sW[bufi][(waveM * 64 + mi * 16 + l16) * WS + (hlf ? 8u : 0u)];
            h8 lo = *(const h8*)(pw);
            h8 hi = *(const h8*)(pw + 16);
            afr[mi] = __builtin_shufflevector(lo, hi, 0,1,2,3,4,5,6,7,8,9,10,11,12,13,14,15);
        }
        // B fragments (32 cin x 16 pixels): lanes 0-15 hold K 0..7 & 16..23, lanes 16-31 K 8..15 & 24..31.
        v16h bfr[2];
#pragma unroll
        for (int ni = 0; ni < 2; ++ni) {
            const _Float16* px = &sX[bufi][(waveN * 32 + ni * 16 + l16) * XS + (hlf ? 8u : 0u)];
            h8 lo = *(const h8*)(px);
            h8 hi = *(const h8*)(px + 16);
            bfr[ni] = __builtin_shufflevector(lo, hi, 0,1,2,3,4,5,6,7,8,9,10,11,12,13,14,15);
        }
#pragma unroll
        for (int mi = 0; mi < 4; ++mi)
#pragma unroll
            for (int ni = 0; ni < 2; ++ni)
                acc[mi][ni] = __builtin_amdgcn_wmma_f32_16x16x32_f16(
                    false, afr[mi], false, bfr[ni],
                    (short)0, acc[mi][ni], false, false);
    };

#if HAVE_ASYNC
    auto stage_async = [&](int k, int bufi) {
        const _Float16 *gx, *gw;
        src_addrs(k, gx, gw);
        _Float16* lx = &sX[bufi][ldsoff];
        _Float16* lw = &sW[bufi][ldsoff];
        async_copy_b128(gx,     lx);
        async_copy_b128(gx + 8, lx + 8);
        async_copy_b128(gw,     lw);
        async_copy_b128(gw + 8, lw + 8);
    };

    stage_async(0, 0);
#pragma unroll 1
    for (int k = 0; k < NK; ++k) {
        const int cur = k & 1;
        if (k + 1 < NK) {
            stage_async(k + 1, cur ^ 1);
            wait_asynccnt<4>();    // in-order completion: stage k has landed
        } else {
            wait_asynccnt<0>();
        }
        __syncthreads();
        compute(cur);
        __syncthreads();           // everyone done reading buf[cur] before overwrite
    }
#else
    auto stage_sync = [&](int k, int bufi) {
        const _Float16 *gx, *gw;
        src_addrs(k, gx, gw);
        float4 a0 = *(const float4*)(gx);
        float4 a1 = *(const float4*)(gx + 8);
        float4 b0 = *(const float4*)(gw);
        float4 b1 = *(const float4*)(gw + 8);
        _Float16* lx = &sX[bufi][ldsoff];
        _Float16* lw = &sW[bufi][ldsoff];
        *(float4*)(lx)     = a0;
        *(float4*)(lx + 8) = a1;
        *(float4*)(lw)     = b0;
        *(float4*)(lw + 8) = b1;
    };

#pragma unroll 1
    for (int k = 0; k < NK; ++k) {
        __syncthreads();
        stage_sync(k, k & 1);
        __syncthreads();
        compute(k & 1);
    }
#endif

    // epilogue: D layout — VGPR r: lanes0-15 (M=r, N=l16), lanes16-31 (M=r+8, N=l16)
#pragma unroll
    for (int ni = 0; ni < 2; ++ni) {
        const unsigned gp2 = pixBase + waveN * 32 + ni * 16 + l16;
        const unsigned b2  = gp2 / HW;
        const unsigned pp2 = gp2 % HW;
        float* obase = out + (size_t)b2 * COUT * HW + pp2;
#pragma unroll
        for (int mi = 0; mi < 4; ++mi) {
            const unsigned cb = coutBase + waveM * 64 + mi * 16 + hlf * 8;
#pragma unroll
            for (int r = 0; r < 8; ++r)
                obase[(size_t)(cb + r) * HW] = acc[mi][ni][r];
        }
    }
}

// ---------------- kernel 4: bias add + output BFP quantize (in-place) ----------------
__global__ void __launch_bounds__(256)
bias_quant_out_kernel(float* __restrict__ out, const float* __restrict__ bias) {
    unsigned id = blockIdx.x * blockDim.x + threadIdx.x;   // over B_*G_*HW
    unsigned p  = id % HW;
    unsigned bg = id / HW;
    unsigned g  = bg % G_;
    unsigned b  = bg / G_;

    float* base = out + (size_t)(b * COUT + g * SZ_) * HW + p;
    float v[SZ_];
    float amax = 0.0f;
#pragma unroll
    for (int c = 0; c < SZ_; ++c) {
        v[c] = base[(size_t)c * HW] + bias[g * SZ_ + c];
        amax = fmaxf(amax, fabsf(v[c]));
    }
    if (amax > 0.0f) {
        int   e      = bfp_exp(amax);
        float scale  = ldexpf(1.0f, e - (QBIT - 1));
        float iscale = ldexpf(1.0f, (QBIT - 1) - e);
#pragma unroll
        for (int c = 0; c < SZ_; ++c) v[c] = bfp_quant_val(v[c], iscale, scale);
    }
#pragma unroll
    for (int c = 0; c < SZ_; ++c) base[(size_t)c * HW] = v[c];
}

// ---------------- launcher ----------------
extern "C" void kernel_launch(void* const* d_in, const int* in_sizes, int n_in,
                              void* d_out, int out_size, void* d_ws, size_t ws_size,
                              hipStream_t stream) {
    (void)in_sizes; (void)n_in; (void)out_size; (void)ws_size;
    const float* x    = (const float*)d_in[0];
    const float* w    = (const float*)d_in[1];
    const float* bias = (const float*)d_in[2];
    float*       out  = (float*)d_out;

    const size_t xq_elems = (size_t)B_ * HP * WP * CIN;          // padded NHWC f16
    const size_t xq_bytes = (xq_elems * sizeof(_Float16) + 255) & ~(size_t)255;
    _Float16* xq = (_Float16*)d_ws;
    _Float16* wq = (_Float16*)((char*)d_ws + xq_bytes);          // (tap,cout,cin) f16

    // zero padded activation buffer (materializes the conv's zero padding)
    (void)hipMemsetAsync(xq, 0, xq_elems * sizeof(_Float16), stream);

    quant_x_kernel<<<(B_ * G_ * HW) / 256, 256, 0, stream>>>(x, xq);
    quant_w_kernel<<<(COUT * 9 * G_ + 255) / 256, 256, 0, stream>>>(w, wq);

    dim3 grid(NPIX / NTP, COUT / MT);   // 784 x 2 blocks
    conv_wmma_kernel<<<grid, 256, 0, stream>>>(xq, wq, out);

    bias_quant_out_kernel<<<(B_ * G_ * HW) / 256, 256, 0, stream>>>(out, bias);
}
